// L4MoE_24850680775019
// MI455X (gfx1250) — matrix operand order repaired and checked by reading, not compiled, not run on hardware
//
#include <hip/hip_runtime.h>
#include <hip/hip_bf16.h>
#include <math.h>

// dims
#define TT 256      // tokens (B*T)
#define CC 1024     // d_model
#define HH 2048     // expert hidden
#define EE 8        // experts

typedef __attribute__((ext_vector_type(16))) __bf16 v16bf;
typedef __attribute__((ext_vector_type(8)))  float  v8f;

// Raw fp32 staging for one 16-element bf16 fragment (4 x float4, distinct regs)
struct Raw16 { float4 q0, q1, q2, q3; };

// A fragment source: lane (l = r + 16*half) holds row M=r.
// elements 0..15 -> K = k0 + 16*half + (0..15)   (chunks at +0, +4, +8, +12)
__device__ __forceinline__ Raw16 load_a_raw(const float* __restrict__ p) {
    Raw16 r;
    r.q0 = *(const float4*)(p);
    r.q1 = *(const float4*)(p + 4);
    r.q2 = *(const float4*)(p + 8);
    r.q3 = *(const float4*)(p + 12);
    return r;
}

// B fragment source: lane (l = n + 16*half) holds column N=n,
// elements 0..15 -> K = k0 + 16*half + (0..15): one contiguous 16-float run.
__device__ __forceinline__ Raw16 load_b_raw(const float* __restrict__ p) {
    Raw16 r;
    r.q0 = *(const float4*)(p);
    r.q1 = *(const float4*)(p + 4);
    r.q2 = *(const float4*)(p + 8);
    r.q3 = *(const float4*)(p + 12);
    return r;
}

__device__ __forceinline__ v16bf cvt_frag(const Raw16& r) {
    v16bf f;
    f[0]  = (__bf16)r.q0.x; f[1]  = (__bf16)r.q0.y; f[2]  = (__bf16)r.q0.z; f[3]  = (__bf16)r.q0.w;
    f[4]  = (__bf16)r.q1.x; f[5]  = (__bf16)r.q1.y; f[6]  = (__bf16)r.q1.z; f[7]  = (__bf16)r.q1.w;
    f[8]  = (__bf16)r.q2.x; f[9]  = (__bf16)r.q2.y; f[10] = (__bf16)r.q2.z; f[11] = (__bf16)r.q2.w;
    f[12] = (__bf16)r.q3.x; f[13] = (__bf16)r.q3.y; f[14] = (__bf16)r.q3.z; f[15] = (__bf16)r.q3.w;
    return f;
}

__device__ __forceinline__ float fast_silu(float g) {
    // g * sigmoid(g) with v_rcp_f32 instead of an IEEE divide
    return g * __builtin_amdgcn_rcpf(1.0f + __expf(-g));
}

// ---------------------------------------------------------------------------
// Router: top-1 argmax + deterministic per-expert token permutation.
__global__ void __launch_bounds__(TT)
moe_router_kernel(const float* __restrict__ x, const float* __restrict__ router,
                  int* __restrict__ counts, int* __restrict__ perm) {
    __shared__ int sidx[TT];
    const int t = threadIdx.x;
    const float* xr = x + (size_t)t * CC;
    float acc[EE];
#pragma unroll
    for (int e = 0; e < EE; ++e) acc[e] = 0.0f;
    for (int k = 0; k < CC; k += 4) {
        float4 xv = *(const float4*)(xr + k);
#pragma unroll
        for (int e = 0; e < EE; ++e) {
            float4 rv = *(const float4*)(router + (size_t)e * CC + k);
            acc[e] += xv.x * rv.x + xv.y * rv.y + xv.z * rv.z + xv.w * rv.w;
        }
    }
    int best = 0;
    float bv = acc[0];
#pragma unroll
    for (int e = 1; e < EE; ++e) {
        if (acc[e] > bv) { bv = acc[e]; best = e; }   // first-max wins (jnp.argmax)
    }
    sidx[t] = best;
    if (t < EE) counts[t] = 0;
    __syncthreads();
    int pos = 0;
    for (int tt = 0; tt < t; ++tt) pos += (sidx[tt] == best) ? 1 : 0;
    perm[best * TT + pos] = t;
    atomicAdd(&counts[best], 1);
}

// ---------------------------------------------------------------------------
// Fused up+gate GEMM + SwiGLU. 8 waves/block; each wave owns TWO 16x16 tiles
// (cols n0, n0+128): A fragment reused across 4 WMMAs per K-step.
// Load phase and convert phase are separated so the 20 b128 loads issue as one
// burst (no per-load s_wait_loadcnt 0).
__global__ void __launch_bounds__(256)
moe_upgate_kernel(const float* __restrict__ x,
                  const float* __restrict__ Wup,
                  const float* __restrict__ Wgate,
                  const int* __restrict__ counts,
                  const int* __restrict__ perm,
                  float* __restrict__ hbuf,
                  int shared_mode, int K) {
    const int e  = blockIdx.z;
    const int m0 = blockIdx.y * 16;
    const int cnt = shared_mode ? TT : counts[e];
    if (m0 >= cnt) return;

    __shared__ int tok[16];
    const int tid = threadIdx.x;
    if (tid < 16) {
        int m = m0 + tid;
        if (m >= cnt) m = cnt - 1;                 // clamp padding rows
        tok[tid] = shared_mode ? (m0 + tid < TT ? m0 + tid : 0)
                               : perm[e * TT + m];
    }
    __syncthreads();

    const int lane = tid & 31;
    const int wv   = tid >> 5;
    const int half = lane >> 4;
    const int l16  = lane & 15;
    const int n0   = blockIdx.x * 256 + wv * 16;   // first column tile in H

    // pre-offset pointers by the per-lane fragment base
    const float* arow  = x + (size_t)tok[l16] * (size_t)K + half * 16;
    const size_t wbase = shared_mode ? 0 : (size_t)e * HH * CC;
    const size_t roff  = (size_t)(n0 + l16) * (size_t)K + half * 16;
    const size_t roff2 = roff + (size_t)128 * (size_t)K;   // second tile: +128 cols
    const float* urow0 = Wup   + wbase + roff;
    const float* urow1 = Wup   + wbase + roff2;
    const float* grow0 = Wgate + wbase + roff;
    const float* grow1 = Wgate + wbase + roff2;

    v8f uacc0 = {}, uacc1 = {};
    v8f gacc0 = {}, gacc1 = {};
    for (int k0 = 0; k0 < K; k0 += 32) {
        __builtin_prefetch(urow0 + k0 + 64, 0, 1);   // speculative stream hints
        __builtin_prefetch(urow1 + k0 + 64, 0, 1);
        __builtin_prefetch(grow0 + k0 + 64, 0, 1);
        __builtin_prefetch(grow1 + k0 + 64, 0, 1);
        // ---- load burst (distinct staging registers) ----
        Raw16 ra  = load_a_raw(arow  + k0);
        Raw16 ru0 = load_b_raw(urow0 + k0);
        Raw16 ru1 = load_b_raw(urow1 + k0);
        Raw16 rg0 = load_b_raw(grow0 + k0);
        Raw16 rg1 = load_b_raw(grow1 + k0);
        // ---- convert phase ----
        v16bf af  = cvt_frag(ra);
        v16bf uf0 = cvt_frag(ru0);
        v16bf uf1 = cvt_frag(ru1);
        v16bf gf0 = cvt_frag(rg0);
        v16bf gf1 = cvt_frag(rg1);
        // ---- matrix phase ----
        uacc0 = __builtin_amdgcn_wmma_f32_16x16x32_bf16(
            false, af, false, uf0, (short)0, uacc0, false, false);
        uacc1 = __builtin_amdgcn_wmma_f32_16x16x32_bf16(
            false, af, false, uf1, (short)0, uacc1, false, false);
        gacc0 = __builtin_amdgcn_wmma_f32_16x16x32_bf16(
            false, af, false, gf0, (short)0, gacc0, false, false);
        gacc1 = __builtin_amdgcn_wmma_f32_16x16x32_bf16(
            false, af, false, gf1, (short)0, gacc1, false, false);
    }

    // h = silu(g) * u; C/D layout: lane holds column N=l16, rows half*8 + i.
#pragma unroll
    for (int i = 0; i < 8; ++i) {
        const int m = half * 8 + i;
        if (m0 + m < cnt) {
            float* hrow = hbuf + (size_t)tok[m] * HH;
            hrow[n0 + l16]       = fast_silu(gacc0[i]) * uacc0[i];
            hrow[n0 + 128 + l16] = fast_silu(gacc1[i]) * uacc1[i];
        }
    }
}

// ---------------------------------------------------------------------------
// Down projection: y = h @ down_e^T, scattered to out by token id.
__global__ void __launch_bounds__(256)
moe_down_kernel(const float* __restrict__ hbuf,
                const float* __restrict__ Wdown,
                const int* __restrict__ counts,
                const int* __restrict__ perm,
                float* __restrict__ out,
                int shared_mode, int K) {
    const int e  = blockIdx.z;
    const int m0 = blockIdx.y * 16;
    const int cnt = shared_mode ? TT : counts[e];
    if (m0 >= cnt) return;

    __shared__ int tok[16];
    const int tid = threadIdx.x;
    if (tid < 16) {
        int m = m0 + tid;
        if (m >= cnt) m = cnt - 1;
        tok[tid] = shared_mode ? (m0 + tid < TT ? m0 + tid : 0)
                               : perm[e * TT + m];
    }
    __syncthreads();

    const int lane = tid & 31;
    const int wv   = tid >> 5;
    const int half = lane >> 4;
    const int l16  = lane & 15;
    const int n0   = blockIdx.x * 256 + wv * 16;   // first column tile in C

    const float* arow  = hbuf + (size_t)tok[l16] * (size_t)K + half * 16;
    const size_t wbase = shared_mode ? 0 : (size_t)e * CC * HH;
    const float* drow0 = Wdown + wbase + (size_t)(n0 + l16) * (size_t)K + half * 16;
    const float* drow1 = drow0 + (size_t)128 * (size_t)K;

    v8f acc0 = {}, acc1 = {};
    for (int k0 = 0; k0 < K; k0 += 32) {
        __builtin_prefetch(drow0 + k0 + 64, 0, 1);
        __builtin_prefetch(drow1 + k0 + 64, 0, 1);
        // ---- load burst ----
        Raw16 ra  = load_a_raw(arow  + k0);
        Raw16 rb0 = load_b_raw(drow0 + k0);
        Raw16 rb1 = load_b_raw(drow1 + k0);
        // ---- convert phase ----
        v16bf af  = cvt_frag(ra);
        v16bf bf0 = cvt_frag(rb0);
        v16bf bf1 = cvt_frag(rb1);
        // ---- matrix phase ----
        acc0 = __builtin_amdgcn_wmma_f32_16x16x32_bf16(
            false, af, false, bf0, (short)0, acc0, false, false);
        acc1 = __builtin_amdgcn_wmma_f32_16x16x32_bf16(
            false, af, false, bf1, (short)0, acc1, false, false);
    }

#pragma unroll
    for (int i = 0; i < 8; ++i) {
        const int m = half * 8 + i;
        if (m0 + m < cnt) {
            float* orow = out + (size_t)tok[m] * CC;
            if (shared_mode) {                     // after expert pass: accumulate
                orow[n0 + l16]       += acc0[i];
                orow[n0 + 128 + l16] += acc1[i];
            } else {                               // tokens partitioned: 1 writer
                orow[n0 + l16]       = acc0[i];
                orow[n0 + 128 + l16] = acc1[i];
            }
        }
    }
}

// ---------------------------------------------------------------------------
extern "C" void kernel_launch(void* const* d_in, const int* in_sizes, int n_in,
                              void* d_out, int out_size, void* d_ws, size_t ws_size,
                              hipStream_t stream) {
    (void)in_sizes; (void)n_in; (void)out_size; (void)ws_size;

    const float* x        = (const float*)d_in[0];   // [256,1024]
    const float* up       = (const float*)d_in[1];   // [8,2048,1024]
    const float* gate     = (const float*)d_in[2];   // [8,2048,1024]
    const float* down     = (const float*)d_in[3];   // [8,1024,2048]
    const float* router   = (const float*)d_in[4];   // [8,1024]
    const float* w_up_s   = (const float*)d_in[5];   // [2048,1024]
    const float* w_gate_s = (const float*)d_in[6];   // [2048,1024]
    const float* w_down_s = (const float*)d_in[7];   // [1024,2048]
    float* out            = (float*)d_out;           // [256,1024]

    // workspace layout (16B-aligned sections)
    char* ws   = (char*)d_ws;
    int*  counts = (int*)ws;                         // 8 ints
    int*  perm   = (int*)(ws + 256);                 // 8*256 ints
    float* h_e   = (float*)(ws + 256 + EE * TT * sizeof(int)); // [256,2048] f32
    float* h_s   = h_e + (size_t)TT * HH;                      // [256,2048] f32

    // 1) route tokens (deterministic perm)
    moe_router_kernel<<<1, TT, 0, stream>>>(x, router, counts, perm);

    // 2) routed experts: up+gate+SwiGLU  (idle tiles exit immediately)
    moe_upgate_kernel<<<dim3(HH / 256, TT / 16, EE), 256, 0, stream>>>(
        x, up, gate, counts, perm, h_e, 0, CC);

    // 3) shared expert: up+gate+SwiGLU over all tokens
    moe_upgate_kernel<<<dim3(HH / 256, TT / 16, 1), 256, 0, stream>>>(
        x, w_up_s, w_gate_s, nullptr, nullptr, h_s, 1, CC);

    // 4) routed experts: down projection, scatter-store y
    moe_down_kernel<<<dim3(CC / 256, TT / 16, EE), 256, 0, stream>>>(
        h_e, down, counts, perm, out, 0, HH);

    // 5) shared expert: down projection, accumulate into out
    moe_down_kernel<<<dim3(CC / 256, TT / 16, 1), 256, 0, stream>>>(
        h_s, w_down_s, nullptr, nullptr, out, 1, HH);
}
